// BDLModel_55241869361629
// MI455X (gfx1250) — matrix operand order repaired and checked by reading, not compiled, not run on hardware
//
#include <hip/hip_runtime.h>
#include <math.h>

#define N_NODES 30000
#define N_EDGES 480000
#define HID 256
#define NB 128
#define SW 512
#define NSAGE 5
#define NL 2
#define OUT_DIM 5
#define EPS_LN 1e-5f

typedef __attribute__((ext_vector_type(16))) _Float16 v16h;
typedef __attribute__((ext_vector_type(8)))  _Float16 v8h;
typedef __attribute__((ext_vector_type(8)))  float    v8f;

__device__ __forceinline__ float gelu_exact(float x) {
  return 0.5f * x * (1.0f + erff(x * 0.70710678118654752440f));
}
__device__ __forceinline__ int imin(int a, int b) { return a < b ? a : b; }

// ---------------------------------------------------------------------------
// WMMA GEMM: C[M,Nc] = epilogue(A[M,K] @ B[K,Nc])
//   A  : f16 row-major [M,K], stride lda
//   BT : f16 row-major [Nc,K] (pre-transposed weights), stride ldbt (=K)
//   C32/C16: either may be null; independent strides.
// flags: bit0 = GELU (after bias), bit1 = residual add (f32 res).
// Block tile 128x128, 8 waves (2Mx4N), wave tile 64x32 = 4x2 WMMA frags
// (8 WMMA / K-step, 12 ds_load_b128 -> 1.5 loads/WMMA, B frag reused 4x).
// Double-buffered LDS, branchless clamped staging, prefetch of tile k+2.
// ---------------------------------------------------------------------------
__global__ __launch_bounds__(256) void gemm_wmma_kernel(
    const _Float16* __restrict__ A, int lda,
    const _Float16* __restrict__ BT, int ldbt,
    const float* __restrict__ bias,
    const float* __restrict__ res, int ldres,
    float* __restrict__ C32, int ldc32,
    _Float16* __restrict__ C16, int ldc16,
    int M, int K, int flags)
{
  __shared__ __align__(16) _Float16 As[2][128][40];  // [m][k], +8 pad
  __shared__ __align__(16) _Float16 Bs[2][128][40];  // [n][k], +8 pad

  const int tid    = threadIdx.x;
  const int blockM = blockIdx.x * 128;
  const int blockN = blockIdx.y * 128;

  const int lane = tid & 31;
  const int wv   = tid >> 5;
  const int wm   = (wv & 1) * 64;   // wave M offset (2 waves down)
  const int wn   = (wv >> 1) * 32;  // wave N offset (4 waves across)
  const int r16  = lane & 15;
  const int hi   = lane >> 4;

  // staging: 16-byte chunks; A rows sr & sr+64, B rows sr & sr+64
  const int sr = tid >> 2;
  const int sc = (tid & 3) << 3;
  const int ra0 = imin(blockM + sr,      M - 1);   // clamped (harmless rows)
  const int ra1 = imin(blockM + sr + 64, M - 1);
  const int rb0 = blockN + sr;
  const int rb1 = blockN + sr + 64;

  v8f acc[4][2] = {};
  const int nt = K >> 5;

  uint4 a0, a1, b0, b1;
  {
    a0 = *(const uint4*)(A  + (size_t)ra0 * lda  + sc);
    a1 = *(const uint4*)(A  + (size_t)ra1 * lda  + sc);
    b0 = *(const uint4*)(BT + (size_t)rb0 * ldbt + sc);
    b1 = *(const uint4*)(BT + (size_t)rb1 * ldbt + sc);
    *(uint4*)(&As[0][sr][sc])      = a0;
    *(uint4*)(&As[0][sr + 64][sc]) = a1;
    *(uint4*)(&Bs[0][sr][sc])      = b0;
    *(uint4*)(&Bs[0][sr + 64][sc]) = b1;
  }
  __syncthreads();

  for (int kt = 0; kt < nt; ++kt) {
    const int cur = kt & 1, nxt = cur ^ 1;

    if (kt + 1 < nt) {               // issue next tile's global loads early
      const int kk = (kt + 1) << 5;
      a0 = *(const uint4*)(A  + (size_t)ra0 * lda  + kk + sc);
      a1 = *(const uint4*)(A  + (size_t)ra1 * lda  + kk + sc);
      b0 = *(const uint4*)(BT + (size_t)rb0 * ldbt + kk + sc);
      b1 = *(const uint4*)(BT + (size_t)rb1 * ldbt + kk + sc);
    }
    if (kt + 2 < nt) {               // L2 prefetch of tile k+2
      const int kk = (kt + 2) << 5;
      __builtin_prefetch(A  + (size_t)ra0 * lda  + kk + sc, 0, 1);
      __builtin_prefetch(BT + (size_t)rb0 * ldbt + kk + sc, 0, 1);
    }

    // A frag (16x32): lane<16 -> M=lane, K 0-7 & 16-23; lane>=16 -> K 8-15 & 24-31
    // B frag (32x16): lane<16 -> N=lane, K 0-7 & 16-23; lane>=16 -> N=lane-16, K 8-15 & 24-31
    union V16 { v16h v; v8h h[2]; };
    V16 af[4], bf[2];
    #pragma unroll
    for (int mi = 0; mi < 4; ++mi) {
      const _Float16* pa = &As[cur][wm + mi * 16 + r16][hi * 8];
      af[mi].h[0] = *(const v8h*)pa;
      af[mi].h[1] = *(const v8h*)(pa + 16);
    }
    #pragma unroll
    for (int ni = 0; ni < 2; ++ni) {
      const _Float16* pb = &Bs[cur][wn + ni * 16 + r16][hi * 8];
      bf[ni].h[0] = *(const v8h*)pb;
      bf[ni].h[1] = *(const v8h*)(pb + 16);
    }
    #pragma unroll
    for (int mi = 0; mi < 4; ++mi)
      #pragma unroll
      for (int ni = 0; ni < 2; ++ni)
        acc[mi][ni] = __builtin_amdgcn_wmma_f32_16x16x32_f16(
            false, af[mi].v, false, bf[ni].v, (short)0, acc[mi][ni], false, false);

    if (kt + 1 < nt) {
      *(uint4*)(&As[nxt][sr][sc])      = a0;
      *(uint4*)(&As[nxt][sr + 64][sc]) = a1;
      *(uint4*)(&Bs[nxt][sr][sc])      = b0;
      *(uint4*)(&Bs[nxt][sr + 64][sc]) = b1;
    }
    __syncthreads();
  }

  // Epilogue. C/D layout: VGPR r -> row = r + 8*hi, col = lane&15.
  #pragma unroll
  for (int mi = 0; mi < 4; ++mi) {
    #pragma unroll
    for (int ni = 0; ni < 2; ++ni) {
      #pragma unroll
      for (int r = 0; r < 8; ++r) {
        int grow = blockM + wm + mi * 16 + r + hi * 8;
        int gcol = blockN + wn + ni * 16 + r16;
        if (grow < M) {
          float v = acc[mi][ni][r];
          if (bias) v += bias[gcol];
          if (flags & 1) v = gelu_exact(v);
          if (flags & 2) v += res[(size_t)grow * ldres + gcol];
          if (C32) C32[(size_t)grow * ldc32 + gcol] = v;
          if (C16) C16[(size_t)grow * ldc16 + gcol] = (_Float16)v;
        }
      }
    }
  }
}

// ---------------------------------------------------------------------------
// weight convert + transpose: W[K][N] f32 -> WT[N][K] f16 (once per call)
__global__ void convT_kernel(const float* __restrict__ W, _Float16* __restrict__ WT,
                             int K, int Nc)
{
  int t = blockIdx.x * blockDim.x + threadIdx.x;
  if (t >= K * Nc) return;
  int k = t / Nc, n = t - k * Nc;
  WT[(size_t)n * K + k] = (_Float16)W[t];
}

__global__ void cvt16_kernel(const float* __restrict__ x, _Float16* __restrict__ o, int n)
{
  int t = blockIdx.x * blockDim.x + threadIdx.x;
  if (t < n) o[t] = (_Float16)x[t];
}

// LayerNorm over HID=256; optional f32 and f16 outputs
__global__ __launch_bounds__(256) void ln_kernel(
    const float* __restrict__ x, const float* __restrict__ g,
    const float* __restrict__ bta,
    float* __restrict__ o32, int ld32,
    _Float16* __restrict__ o16, int ld16)
{
  __shared__ float red[256];
  const int row = blockIdx.x, tid = threadIdx.x;
  float v = x[(size_t)row * HID + tid];
  red[tid] = v; __syncthreads();
  #pragma unroll
  for (int s = 128; s > 0; s >>= 1) { if (tid < s) red[tid] += red[tid + s]; __syncthreads(); }
  float mean = red[0] * (1.0f / HID);
  __syncthreads();
  float d = v - mean;
  red[tid] = d * d; __syncthreads();
  #pragma unroll
  for (int s = 128; s > 0; s >>= 1) { if (tid < s) red[tid] += red[tid + s]; __syncthreads(); }
  float var = red[0] * (1.0f / HID);
  float r = d * rsqrtf(var + EPS_LN) * g[tid] + bta[tid];
  if (o32) o32[(size_t)row * ld32 + tid] = r;
  if (o16) o16[(size_t)row * ld16 + tid] = (_Float16)r;
}

// 2x2 Householder closed form; only X[1][0] matters.
__global__ void householder_kernel(const _Float16* __restrict__ nr, float* __restrict__ O, int total)
{
  int t = blockIdx.x * blockDim.x + threadIdx.x;
  if (t >= total) return;
  float a = (float)nr[(size_t)t * 4 + 2];
  float tau0 = 2.0f / (1.0f + a * a);
  float q00 = 1.0f - tau0;
  float q01 = tau0 * a;
  float q10 = -tau0 * a;
  float q11 = -(1.0f - tau0 * a * a);
  *(float4*)(O + (size_t)t * 4) = make_float4(q00, q01, q10, q11);
}

// y[n,b,i] = O[n,b,i,j] * hn[n,b*2+j]  (hn = hc[:, :256] f16, ld 512)
__global__ void rotate_kernel(const float* __restrict__ O, const _Float16* __restrict__ hc,
                              _Float16* __restrict__ y, int total)
{
  int t = blockIdx.x * blockDim.x + threadIdx.x;
  if (t >= total) return;
  int n = t >> 7, bb = t & 127;
  float4 o = *(const float4*)(O + (size_t)t * 4);
  float h0 = (float)hc[(size_t)n * 512 + bb * 2];
  float h1 = (float)hc[(size_t)n * 512 + bb * 2 + 1];
  y[(size_t)n * 256 + bb * 2]     = (_Float16)(o.x * h0 + o.y * h1);
  y[(size_t)n * 256 + bb * 2 + 1] = (_Float16)(o.z * h0 + o.w * h1);
}

// msg[n,b,i] = O[n,b,j,i] * agg[n,b*2+j] -> hc[:, 256:512] f16
__global__ void msg_kernel(const float* __restrict__ O, const float* __restrict__ agg,
                           _Float16* __restrict__ hc, int total)
{
  int t = blockIdx.x * blockDim.x + threadIdx.x;
  if (t >= total) return;
  int n = t >> 7, bb = t & 127;
  float4 o = *(const float4*)(O + (size_t)t * 4);
  float a0 = agg[(size_t)n * 256 + bb * 2];
  float a1 = agg[(size_t)n * 256 + bb * 2 + 1];
  hc[(size_t)n * 512 + 256 + bb * 2]     = (_Float16)(o.x * a0 + o.z * a1);
  hc[(size_t)n * 512 + 256 + bb * 2 + 1] = (_Float16)(o.y * a0 + o.w * a1);
}

__global__ void zero_kernel(float* __restrict__ p, int n)
{
  int t = blockIdx.x * blockDim.x + threadIdx.x;
  if (t < n) p[t] = 0.0f;
}

__global__ void deg_kernel(const int* __restrict__ dst, float* __restrict__ deg, int ne)
{
  int t = blockIdx.x * blockDim.x + threadIdx.x;
  if (t < ne) atomicAdd(&deg[dst[t]], 1.0f);
}

// one block per edge: out[dst] += (float)x[src]  (f16 rows, f32 accumulation)
__global__ __launch_bounds__(256) void edge_scatter_kernel(
    const _Float16* __restrict__ x, int ldx,
    const int* __restrict__ src, const int* __restrict__ dst,
    float* __restrict__ out, int ldo, int W)
{
  int e = blockIdx.x;
  int s = src[e], d = dst[e];
  const _Float16* xs = x + (size_t)s * ldx;
  float* od = out + (size_t)d * ldo;
  for (int c = threadIdx.x; c < W; c += blockDim.x)
    atomicAdd(od + c, (float)xs[c]);
}

// acc f32 -> divide by deg -> f16 out (strided)
__global__ void mean_to_f16_kernel(const float* __restrict__ acc, int ldacc, int W,
                                   const float* __restrict__ deg,
                                   _Float16* __restrict__ o, int ldo, int rows)
{
  int t = blockIdx.x * blockDim.x + threadIdx.x;
  if (t >= rows * W) return;
  int r = t / W, c = t - r * W;
  o[(size_t)r * ldo + c] =
      (_Float16)(acc[(size_t)r * ldacc + c] * (1.0f / fmaxf(deg[r], 1.0f)));
}

__global__ void mean_inplace_kernel(float* __restrict__ acc, int ldacc, int W,
                                    const float* __restrict__ deg, int rows)
{
  int t = blockIdx.x * blockDim.x + threadIdx.x;
  if (t >= rows * W) return;
  int r = t / W, c = t - r * W;
  acc[(size_t)r * ldacc + c] *= 1.0f / fmaxf(deg[r], 1.0f);
}

// final projection: [N,256] @ [256,5] + b
__global__ void out_kernel(const float* __restrict__ hn, const float* __restrict__ w,
                           const float* __restrict__ bo, float* __restrict__ out, int n)
{
  int t = blockIdx.x * blockDim.x + threadIdx.x;
  if (t >= n * 8) return;
  int row = t >> 3, o = t & 7;
  if (o >= OUT_DIM) return;
  float acc = bo[o];
  #pragma unroll 8
  for (int k2 = 0; k2 < HID; ++k2)
    acc += hn[(size_t)row * HID + k2] * w[k2 * OUT_DIM + o];
  out[(size_t)row * OUT_DIM + o] = acc;
}

// ---------------------------------------------------------------------------

static void gemm(hipStream_t s, const _Float16* A, int lda, const _Float16* BT, int ldbt,
                 const float* bias, const float* res, int ldres,
                 float* C32, int ldc32, _Float16* C16, int ldc16,
                 int M, int K, int Nc, int flags)
{
  dim3 grid((M + 127) / 128, Nc / 128);
  gemm_wmma_kernel<<<grid, 256, 0, s>>>(A, lda, BT, ldbt, bias, res, ldres,
                                        C32, ldc32, C16, ldc16, M, K, flags);
}

static void convT(hipStream_t s, const float* W, _Float16* WT, int K, int Nc)
{
  convT_kernel<<<(K * Nc + 255) / 256, 256, 0, s>>>(W, WT, K, Nc);
}

extern "C" void kernel_launch(void* const* d_in, const int* in_sizes, int n_in,
                              void* d_out, int out_size, void* d_ws, size_t ws_size,
                              hipStream_t stream)
{
  const float* x        = (const float*)d_in[0];
  const int*   eidx     = (const int*)d_in[1];
  const float* w_in     = (const float*)d_in[2];
  const float* b_in     = (const float*)d_in[3];
  const float* se_in_w  = (const float*)d_in[4];
  const float* se_in_b  = (const float*)d_in[5];
  const float* sage_w1  = (const float*)d_in[6];
  const float* sage_b1  = (const float*)d_in[7];
  const float* sage_w2  = (const float*)d_in[8];
  const float* sage_b2  = (const float*)d_in[9];
  const float* se_out_w = (const float*)d_in[10];
  const float* se_out_b = (const float*)d_in[11];
  const float* enc_w1   = (const float*)d_in[12];
  const float* enc_b1   = (const float*)d_in[13];
  const float* enc_w2   = (const float*)d_in[14];
  const float* enc_b2   = (const float*)d_in[15];
  const float* ln_g     = (const float*)d_in[16];
  const float* ln_b     = (const float*)d_in[17];
  const float* bdl_w1   = (const float*)d_in[18];
  const float* bdl_b1   = (const float*)d_in[19];
  const float* bdl_w2   = (const float*)d_in[20];
  const float* bdl_b2   = (const float*)d_in[21];
  const float* out_ln_g = (const float*)d_in[22];
  const float* out_ln_b = (const float*)d_in[23];
  const float* w_out    = (const float*)d_in[24];
  const float* b_out    = (const float*)d_in[25];

  const int* src = eidx;
  const int* dst = eidx + N_EDGES;

  // ---- workspace carve (bytes) ----
  char* p = (char*)d_ws;
  auto carve = [&](size_t bytes) { char* q = p; p += (bytes + 255) & ~(size_t)255; return q; };

  float* z32  = (float*)carve((size_t)N_NODES * SW * 4);       // z residual (f32)
  float* aggW = (float*)carve((size_t)N_NODES * SW * 4);       // scatter accumulator
  float* O32  = (float*)carve((size_t)N_NODES * SW * 4);       // bundle frames
  float* h32  = (float*)carve((size_t)N_NODES * HID * 4);      // h residual (f32)
  float* deg  = (float*)carve((size_t)N_NODES * 4);

  _Float16* zch = (_Float16*)carve((size_t)N_NODES * 2 * SW * 2);  // [z | m] f16
  _Float16* t1h = (_Float16*)carve((size_t)N_NODES * SW * 2);
  _Float16* t2h = (_Float16*)carve((size_t)N_NODES * SW * 2);
  _Float16* hch = (_Float16*)carve((size_t)N_NODES * 2 * HID * 2); // [hn | msg] f16
  _Float16* yh  = (_Float16*)carve((size_t)N_NODES * HID * 2);
  _Float16* hh  = (_Float16*)carve((size_t)N_NODES * HID * 2);
  _Float16* xh  = (_Float16*)carve((size_t)N_NODES * HID * 2);

  // transposed f16 weights
  _Float16* w_in_t   = (_Float16*)carve((size_t)256 * 256 * 2);
  _Float16* se_in_t  = (_Float16*)carve((size_t)512 * 256 * 2);
  _Float16* sage_w1t = (_Float16*)carve((size_t)NSAGE * 512 * 1024 * 2);
  _Float16* sage_w2t = (_Float16*)carve((size_t)NSAGE * 512 * 512 * 2);
  _Float16* se_out_t = (_Float16*)carve((size_t)512 * 512 * 2);
  _Float16* enc_w1t  = (_Float16*)carve((size_t)NL * 512 * 512 * 2);
  _Float16* enc_w2t  = (_Float16*)carve((size_t)NL * 512 * 512 * 2);
  _Float16* bdl_w1t  = (_Float16*)carve((size_t)NL * 256 * 512 * 2);
  _Float16* bdl_w2t  = (_Float16*)carve((size_t)NL * 256 * 256 * 2);

  const int NBLK = 256;
  const int nbTot = N_NODES * NB;

  // ---- one-time (per call) conversions ----
  cvt16_kernel<<<(N_NODES * HID + NBLK - 1) / NBLK, NBLK, 0, stream>>>(x, xh, N_NODES * HID);
  convT(stream, w_in, w_in_t, 256, 256);
  convT(stream, se_in_w, se_in_t, 256, 512);
  for (int i = 0; i < NSAGE; ++i) {
    convT(stream, sage_w1 + (size_t)i * 1024 * 512, sage_w1t + (size_t)i * 512 * 1024, 1024, 512);
    convT(stream, sage_w2 + (size_t)i * 512 * 512,  sage_w2t + (size_t)i * 512 * 512,  512, 512);
  }
  convT(stream, se_out_w, se_out_t, 512, 512);
  for (int k = 0; k < NL; ++k) {
    convT(stream, enc_w1 + (size_t)k * 512 * 512, enc_w1t + (size_t)k * 512 * 512, 512, 512);
    convT(stream, enc_w2 + (size_t)k * 512 * 512, enc_w2t + (size_t)k * 512 * 512, 512, 512);
    convT(stream, bdl_w1 + (size_t)k * 512 * 256, bdl_w1t + (size_t)k * 256 * 512, 512, 256);
    convT(stream, bdl_w2 + (size_t)k * 256 * 256, bdl_w2t + (size_t)k * 256 * 256, 256, 256);
  }

  // in-degree
  zero_kernel<<<(N_NODES + NBLK - 1) / NBLK, NBLK, 0, stream>>>(deg, N_NODES);
  deg_kernel<<<(N_EDGES + NBLK - 1) / NBLK, NBLK, 0, stream>>>(dst, deg, N_EDGES);

  // h = gelu(x @ w_in + b_in)   -> h32 + hh
  gemm(stream, xh, HID, w_in_t, HID, b_in, nullptr, 0,
       h32, HID, hh, HID, N_NODES, HID, HID, 1);

  for (int k = 0; k < NL; ++k) {
    // ---- struct encoder ----
    // z = gelu(h @ se_in_w + b) -> z32 + zch[:, :SW]
    gemm(stream, hh, HID, se_in_t, HID, se_in_b, nullptr, 0,
         z32, SW, zch, 2 * SW, N_NODES, HID, SW, 1);
    for (int i = 0; i < NSAGE; ++i) {
      zero_kernel<<<(N_NODES * SW + NBLK - 1) / NBLK, NBLK, 0, stream>>>(aggW, N_NODES * SW);
      edge_scatter_kernel<<<N_EDGES, NBLK, 0, stream>>>(zch, 2 * SW, src, dst, aggW, SW, SW);
      mean_to_f16_kernel<<<(N_NODES * SW + NBLK - 1) / NBLK, NBLK, 0, stream>>>(
          aggW, SW, SW, deg, zch + SW, 2 * SW, N_NODES);
      // t1 = gelu(zc @ sage_w1[i] + b1[i])
      gemm(stream, zch, 2 * SW, sage_w1t + (size_t)i * 512 * 1024, 1024, sage_b1 + i * SW,
           nullptr, 0, nullptr, 0, t1h, SW, N_NODES, 2 * SW, SW, 1);
      // z += t1 @ sage_w2[i] + b2[i]   -> z32 + zch[:, :SW]
      gemm(stream, t1h, SW, sage_w2t + (size_t)i * 512 * 512, 512, sage_b2 + i * SW,
           z32, SW, z32, SW, zch, 2 * SW, N_NODES, SW, SW, 2);
    }
    // enc = z @ se_out_w + b -> t1h
    gemm(stream, zch, 2 * SW, se_out_t, 512, se_out_b, nullptr, 0,
         nullptr, 0, t1h, SW, N_NODES, SW, SW, 0);

    // ---- enc MLP -> nr (t1h) ----
    gemm(stream, t1h, SW, enc_w1t + (size_t)k * 512 * 512, 512, enc_b1 + k * SW,
         nullptr, 0, nullptr, 0, t2h, SW, N_NODES, SW, SW, 1);
    gemm(stream, t2h, SW, enc_w2t + (size_t)k * 512 * 512, 512, enc_b2 + k * SW,
         nullptr, 0, nullptr, 0, t1h, SW, N_NODES, SW, SW, 0);

    // ---- bundle frames + transport ----
    householder_kernel<<<(nbTot + NBLK - 1) / NBLK, NBLK, 0, stream>>>(t1h, O32, nbTot);
    ln_kernel<<<N_NODES, 256, 0, stream>>>(h32, ln_g + k * HID, ln_b + k * HID,
                                           nullptr, 0, hch, 2 * HID);
    rotate_kernel<<<(nbTot + NBLK - 1) / NBLK, NBLK, 0, stream>>>(O32, hch, yh, nbTot);
    zero_kernel<<<(N_NODES * HID + NBLK - 1) / NBLK, NBLK, 0, stream>>>(aggW, N_NODES * HID);
    edge_scatter_kernel<<<N_EDGES, NBLK, 0, stream>>>(yh, HID, src, dst, aggW, HID, HID);
    mean_inplace_kernel<<<(N_NODES * HID + NBLK - 1) / NBLK, NBLK, 0, stream>>>(
        aggW, HID, HID, deg, N_NODES);
    msg_kernel<<<(nbTot + NBLK - 1) / NBLK, NBLK, 0, stream>>>(O32, aggW, hch, nbTot);

    // ---- BDL MLP residual ----
    gemm(stream, hch, 2 * HID, bdl_w1t + (size_t)k * 256 * 512, 512, bdl_b1 + k * HID,
         nullptr, 0, nullptr, 0, yh, HID, N_NODES, 2 * HID, HID, 1);
    gemm(stream, yh, HID, bdl_w2t + (size_t)k * 256 * 256, 256, bdl_b2 + k * HID,
         h32, HID, h32, HID, hh, HID, N_NODES, HID, HID, 2);
  }

  // final LN + projection (f32 path, tiny)
  ln_kernel<<<N_NODES, 256, 0, stream>>>(h32, out_ln_g, out_ln_b, aggW, HID, nullptr, 0);
  out_kernel<<<(N_NODES * 8 + NBLK - 1) / NBLK, NBLK, 0, stream>>>(
      aggW, w_out, b_out, (float*)d_out, N_NODES);
}
